// NTXentLoss_23235773072171
// MI455X (gfx1250) — compile-verified, matrix operand order checked by reading, not run on hardware
//
#include <hip/hip_runtime.h>
#include <hip/hip_bf16.h>

// NT-Xent loss, M = 8192 rows, D = 256, N = 4096, T = 0.5.
// Pipeline: [k0] f32->bf16 pack  [k1] sim = 2*z z^T via v_wmma_f32_16x16x32_bf16
//           (staging via global_load_async_to_lds_b128, ASYNCcnt tracked)
//           [k2] per-row: lse, exact top-2048 threshold (LDS radix select), losses
//           [k3] per-row: count[j] vs thresholds -> weighted term
//           [k4] final mean.
// Workspace: 4MB zbf + 256MB sim + 3*32KB vectors (~260.1 MB required).

#define MDIM 8192
#define NDIM 4096
#define DDIM 256
#define KSEL 2048   // N//2

typedef __attribute__((ext_vector_type(16))) __bf16 v16bf;
typedef __attribute__((ext_vector_type(8)))  float  v8f;

union Frag16 { unsigned u[8]; v16bf v; };

__device__ __forceinline__ unsigned monokey(float x) {
    unsigned u = __float_as_uint(x);
    return (u & 0x80000000u) ? ~u : (u | 0x80000000u);
}
__device__ __forceinline__ float invmonokey(unsigned k) {
    unsigned u = (k & 0x80000000u) ? (k & 0x7fffffffu) : ~k;
    return __uint_as_float(u);
}

// ---------------- kernel 0: pack z = concat(z_i, z_j) into bf16 ----------------
__global__ __launch_bounds__(256) void k0_pack_bf16(const float* __restrict__ zi,
                                                    const float* __restrict__ zj,
                                                    __bf16* __restrict__ zb) {
    int idx = blockIdx.x * 256 + threadIdx.x;      // 0 .. 8192*256-1
    if (idx >= MDIM * DDIM) return;
    float v = (idx < NDIM * DDIM) ? zi[idx] : zj[idx - NDIM * DDIM];
    zb[idx] = (__bf16)v;
}

// ---------------- kernel 1: sim = 2 * z z^T, diagonal = -1e9 -------------------
// Block: 256 threads (8 waves), computes a 128x128 tile. Wave grid 2x4,
// each wave owns 4x2 subtiles of 16x16, K swept in steps of 32.
// Staging: global_load_async_to_lds_b128 (direct global->LDS, no VGPR bounce);
// INST_OFFSET applies to both LDS and global address per CDNA5 ISA.
__global__ __launch_bounds__(256) void k1_sim_gemm(const __bf16* __restrict__ zb,
                                                   float* __restrict__ sim) {
    // LDS rows padded to 17 dwords (34 bf16) -> conflict-free fragment reads.
    __shared__ __align__(16) unsigned shA[128 * 17];
    __shared__ __align__(16) unsigned shB[128 * 17];

    const int tid  = threadIdx.x;
    const int rowBase = blockIdx.y * 128;
    const int colBase = blockIdx.x * 128;
    const int wave = tid >> 5, lane = tid & 31;
    const int wr = wave >> 2;          // 0..1 -> 64 rows
    const int wc = wave & 3;           // 0..3 -> 32 cols
    const int m    = lane & 15;
    const int half = lane >> 4;

    v8f acc[4][2];
#pragma unroll
    for (int r = 0; r < 4; r++)
#pragma unroll
        for (int c = 0; c < 2; c++)
#pragma unroll
            for (int e = 0; e < 8; e++) acc[r][c][e] = 0.0f;

    const int lrow = tid >> 1;                  // 0..127
    const int lcolD = (tid & 1) * 8;            // dword col in LDS (0 or 8)
    const int gcolE = (tid & 1) * 16;           // bf16 col in global

    // LDS byte offsets this thread stages into (generic ptr low 32 bits = LDS offset)
    const unsigned ldsA = (unsigned)(uintptr_t)(&shA[lrow * 17 + lcolD]);
    const unsigned ldsB = (unsigned)(uintptr_t)(&shB[lrow * 17 + lcolD]);

    // k-pair dword index within a 32-wide K slice for fragment vgpr v:
    // k(v) = (v<4 ? 2v : 16+2(v-4)) + half*8  ->  kd = (v<4 ? v : v+4) + half*4
#pragma unroll
    for (int kt = 0; kt < DDIM / 32; kt++) {
        __syncthreads();   // previous compute must be done before LDS overwrite
        {
            const __bf16* gA = zb + (size_t)(rowBase + lrow) * DDIM + kt * 32 + gcolE;
            const __bf16* gB = zb + (size_t)(colBase + lrow) * DDIM + kt * 32 + gcolE;
            // 32B per thread per matrix: two b128 async copies, offset hits both sides
            asm volatile(
                "global_load_async_to_lds_b128 %0, %2, off\n\t"
                "global_load_async_to_lds_b128 %0, %2, off offset:16\n\t"
                "global_load_async_to_lds_b128 %1, %3, off\n\t"
                "global_load_async_to_lds_b128 %1, %3, off offset:16"
                :: "v"(ldsA), "v"(ldsB), "v"(gA), "v"(gB)
                : "memory");
        }
        if (kt + 1 < DDIM / 32) {   // gfx1250 global_prefetch_b8 for the next K slice
            __builtin_prefetch(zb + (size_t)(rowBase + lrow) * DDIM + (kt + 1) * 32 + gcolE, 0, 3);
            __builtin_prefetch(zb + (size_t)(colBase + lrow) * DDIM + (kt + 1) * 32 + gcolE, 0, 3);
        }
        asm volatile("s_wait_asynccnt 0" ::: "memory");
        __syncthreads();

        Frag16 af[4], bf[2];
#pragma unroll
        for (int r = 0; r < 4; r++) {
            int row = wr * 64 + r * 16 + m;
#pragma unroll
            for (int v = 0; v < 8; v++) {
                int kd = (v < 4 ? v : v + 4) + half * 4;
                af[r].u[v] = shA[row * 17 + kd];
            }
        }
#pragma unroll
        for (int c = 0; c < 2; c++) {
            int col = wc * 32 + c * 16 + m;
#pragma unroll
            for (int v = 0; v < 8; v++) {
                int kd = (v < 4 ? v : v + 4) + half * 4;
                bf[c].u[v] = shB[col * 17 + kd];
            }
        }
#pragma unroll
        for (int r = 0; r < 4; r++)
#pragma unroll
            for (int c = 0; c < 2; c++)
                acc[r][c] = __builtin_amdgcn_wmma_f32_16x16x32_bf16(
                    false, af[r].v, false, bf[c].v, (short)0, acc[r][c], false, false);
    }

    // C/D layout: vgpr v -> M = v + 8*half ; N = lane&15
#pragma unroll
    for (int r = 0; r < 4; r++) {
        int gr0 = rowBase + wr * 64 + r * 16;
#pragma unroll
        for (int c = 0; c < 2; c++) {
            int gc = colBase + wc * 32 + c * 16 + m;
#pragma unroll
            for (int v = 0; v < 8; v++) {
                int gr = gr0 + v + half * 8;
                float val = acc[r][c][v] * 2.0f;          // 1/T = 2
                if (gr == gc) val = -1e9f;                // diagonal mask
                sim[(size_t)gr * MDIM + gc] = val;
            }
        }
    }
}

// ---------------- kernel 2: per-row lse, exact radix-select threshold, loss ----
__global__ __launch_bounds__(256) void k2_row_stats(const float* __restrict__ sim,
                                                    float* __restrict__ thresh,
                                                    float* __restrict__ losses) {
    __shared__ __align__(16) float srow[MDIM];
    __shared__ unsigned hist[256];
    __shared__ float sred[256];
    __shared__ unsigned s_prefix;
    __shared__ int s_k;
    __shared__ float s_lse;

    const int j = blockIdx.x, tid = threadIdx.x;
    const float* row = sim + (size_t)j * MDIM;

    const float4* r4 = (const float4*)row;
    float4* s4 = (float4*)srow;
    for (int i = tid; i < MDIM / 4; i += 256) s4[i] = r4[i];
    __syncthreads();

    // logsumexp (sim in [-2,2] except diag -1e9 -> expf underflows to 0; stable as-is)
    float part = 0.0f;
    for (int i = tid; i < MDIM; i += 256) part += expf(srow[i]);
    sred[tid] = part; __syncthreads();
    for (int s = 128; s > 0; s >>= 1) { if (tid < s) sred[tid] += sred[tid + s]; __syncthreads(); }
    if (tid == 0) s_lse = logf(sred[0]);
    __syncthreads();
    const float lse = s_lse;

    // exact MSB-first radix select of the KSEL-th largest value
    if (tid == 0) { s_prefix = 0u; s_k = KSEL; }
    __syncthreads();
#pragma unroll
    for (int pass = 0; pass < 4; pass++) {
        const int shift = 24 - pass * 8;
        const unsigned hiMask = (pass == 0) ? 0u : (0xFFFFFFFFu << (shift + 8));
        hist[tid] = 0u;
        __syncthreads();
        const unsigned pfx = s_prefix;
        for (int i = tid; i < MDIM; i += 256) {
            unsigned key = monokey(srow[i]);
            if ((key & hiMask) == pfx) atomicAdd(&hist[(key >> shift) & 255u], 1u);
        }
        __syncthreads();
        if (tid == 0) {
            int kk = s_k; unsigned cum = 0; int chosen = 0;
            for (int b = 255; b >= 0; b--) {
                unsigned h = hist[b];
                if (cum + h >= (unsigned)kk) { chosen = b; s_k = kk - (int)cum; break; }
                cum += h;
            }
            s_prefix = pfx | ((unsigned)chosen << shift);
        }
        __syncthreads();
    }
    if (tid == 0) thresh[j] = invmonokey(s_prefix);

    // second softmax denominator: sum_k exp(p_k), p_k = exp(sim - lse)
    float p2 = 0.0f;
    for (int i = tid; i < MDIM; i += 256) p2 += expf(expf(srow[i] - lse));
    sred[tid] = p2; __syncthreads();
    for (int s = 128; s > 0; s >>= 1) { if (tid < s) sred[tid] += sred[tid + s]; __syncthreads(); }
    if (tid == 0) {
        int label = (j < NDIM) ? (j + NDIM) : (j - NDIM);
        float pl = expf(srow[label] - lse);
        losses[j] = logf(sred[0]) - pl;     // -log_softmax(p)[label]
    }
}

// ---------------- kernel 3: count[j] vs thresholds -> weighted term -----------
__global__ __launch_bounds__(256) void k3_col_weight(const float* __restrict__ sim,
                                                     const float* __restrict__ thresh,
                                                     const float* __restrict__ losses,
                                                     float* __restrict__ terms) {
    __shared__ int ired[256];
    const int j = blockIdx.x, tid = threadIdx.x;
    const float* row = sim + (size_t)j * MDIM;   // sim is symmetric: row j == column j
    int cnt = 0;
    for (int i = tid; i < MDIM; i += 256)
        cnt += (row[i] >= thresh[i]) ? 1 : 0;    // i==j auto-excluded (-1e9)
    ired[tid] = cnt; __syncthreads();
    for (int s = 128; s > 0; s >>= 1) { if (tid < s) ired[tid] += ired[tid + s]; __syncthreads(); }
    if (tid == 0) {
        float colw = 0.3f * (float)MDIM + 0.7f + 0.4f * (float)ired[0];
        if (j < NDIM) {
            // override w[N+j, j] = 1.0 ; tk(N+j, j) uses sim[N+j, j] == row_j[j+N]
            float sjl = row[j + NDIM];
            colw += 0.7f - ((sjl >= thresh[j + NDIM]) ? 0.4f : 0.0f);
        }
        terms[j] = losses[j] * colw;
    }
}

// ---------------- kernel 4: final mean -----------------------------------------
__global__ __launch_bounds__(256) void k4_reduce(const float* __restrict__ terms,
                                                 float* __restrict__ out) {
    __shared__ float sred[256];
    const int tid = threadIdx.x;
    float p = 0.0f;
    for (int i = tid; i < MDIM; i += 256) p += terms[i];
    sred[tid] = p; __syncthreads();
    for (int s = 128; s > 0; s >>= 1) { if (tid < s) sred[tid] += sred[tid + s]; __syncthreads(); }
    if (tid == 0) out[0] = sred[0] / ((float)MDIM * (float)MDIM);
}

extern "C" void kernel_launch(void* const* d_in, const int* in_sizes, int n_in,
                              void* d_out, int out_size, void* d_ws, size_t ws_size,
                              hipStream_t stream) {
    const float* zi = (const float*)d_in[0];
    const float* zj = (const float*)d_in[1];
    float* out = (float*)d_out;

    char* ws = (char*)d_ws;
    __bf16* zb   = (__bf16*)ws;                                   // 4 MB
    float*  sim  = (float*)(ws + (size_t)(4 << 20));              // 256 MB
    float*  thr  = (float*)(ws + (size_t)(4 << 20) + ((size_t)MDIM * MDIM * 4));
    float*  loss = thr + MDIM;
    float*  term = loss + MDIM;

    k0_pack_bf16<<<(MDIM * DDIM) / 256, 256, 0, stream>>>(zi, zj, zb);
    dim3 g1(MDIM / 128, MDIM / 128);
    k1_sim_gemm<<<g1, 256, 0, stream>>>(zb, sim);
    k2_row_stats<<<MDIM, 256, 0, stream>>>(sim, thr, loss);
    k3_col_weight<<<MDIM, 256, 0, stream>>>(sim, thr, loss, term);
    k4_reduce<<<1, 256, 0, stream>>>(term, out);
}
